// MultiHeadSelfAttention_1924145348881
// MI455X (gfx1250) — matrix operand order repaired and checked by reading, not compiled, not run on hardware
//
#include <hip/hip_runtime.h>
#include <hip/hip_bf16.h>

typedef __attribute__((ext_vector_type(16))) __bf16 v16bf;
typedef __attribute__((ext_vector_type(8)))  __bf16 v8bf;
typedef __attribute__((ext_vector_type(8)))  float  v8f;

#define EMBED 1024
#define NHEAD 16
#define HDIM  64
#define SEQ   2048
#define BATCH 2
#define MROWS (BATCH*SEQ)   /* 4096 */

// Assemble a 16-half WMMA operand from two 16-byte chunks.
__device__ __forceinline__ v16bf load_a16(const __bf16* p0, const __bf16* p1) {
    v16bf a;
    ((v8bf*)&a)[0] = *(const v8bf*)p0;
    ((v8bf*)&a)[1] = *(const v8bf*)p1;
    return a;
}

__device__ __forceinline__ v8f wmma_bf16(v16bf a, v16bf b, v8f c) {
    return __builtin_amdgcn_wmma_f32_16x16x32_bf16(false, a, false, b, (short)0, c, false, false);
}

// ---------------- prep kernels ----------------

__global__ void cvt_f32_bf16(const float* __restrict__ src, __bf16* __restrict__ dst, int n) {
    int i = blockIdx.x * blockDim.x + threadIdx.x;
    if (i < n) dst[i] = (__bf16)src[i];
}

// W: [K=EMBED, N=EMBED] row-major fp32 -> Wt: [N, K] bf16 (transposed).
// LDS-tiled 32x32 transpose: coalesced global reads AND writes.
__global__ void transpose_cvt(const float* __restrict__ W, __bf16* __restrict__ Wt) {
    __shared__ __bf16 tile[32][33];            // pad to avoid bank conflicts
    const int tx = threadIdx.x, ty = threadIdx.y;
    const int n = blockIdx.x * 32 + tx;        // column of W (fast axis, coalesced read)
    const int k = blockIdx.y * 32 + ty;        // row of W
#pragma unroll
    for (int j = 0; j < 32; j += 8)
        tile[ty + j][tx] = (__bf16)W[(k + j) * EMBED + n];
    __syncthreads();
    const int ko = blockIdx.y * 32 + tx;       // column of Wt (fast axis, coalesced write)
    const int no = blockIdx.x * 32 + ty;       // row of Wt
#pragma unroll
    for (int j = 0; j < 32; j += 8)
        Wt[(no + j) * EMBED + ko] = tile[tx][ty + j];
}

// ---------------- shared GEMM core: one wave computes 32x64 of A[M,K] @ Wt^T ----------------
// A row-major bf16 [*,EMBED], Wt stored [N,K]. Two-stage register double buffering.

__device__ __forceinline__ void gemm_load_step(const __bf16* __restrict__ A,
                                               const __bf16* __restrict__ Wt,
                                               int m0, int n0, int k0, int ln, int half,
                                               v16bf a[2], v16bf b[4]) {
#pragma unroll
    for (int mt = 0; mt < 2; ++mt) {
        const __bf16* ap = A + (m0 + mt * 16 + ln) * EMBED + k0 + (half ? 8 : 0);
        a[mt] = load_a16(ap, ap + 16);
    }
#pragma unroll
    for (int nt = 0; nt < 4; ++nt) {
        const __bf16* bp = Wt + (n0 + nt * 16 + ln) * EMBED + k0 + (half ? 8 : 0);
        b[nt] = load_a16(bp, bp + 16);
    }
}

__device__ __forceinline__ void gemm_32x64(const __bf16* __restrict__ A,
                                           const __bf16* __restrict__ Wt,
                                           int m0, int n0, int ln, int half,
                                           v8f acc[2][4]) {
    v16bf a0[2], b0[4], a1[2], b1[4];
    gemm_load_step(A, Wt, m0, n0, 0, ln, half, a0, b0);
#pragma unroll 1
    for (int k0 = 0; k0 < EMBED; k0 += 64) {
        // prefetch k0+32, then compute with (a0,b0)
        gemm_load_step(A, Wt, m0, n0, k0 + 32, ln, half, a1, b1);
#pragma unroll
        for (int mt = 0; mt < 2; ++mt)
#pragma unroll
            for (int nt = 0; nt < 4; ++nt)
                acc[mt][nt] = wmma_bf16(a0[mt], b0[nt], acc[mt][nt]);
        // prefetch k0+64, then compute with (a1,b1)
        if (k0 + 64 < EMBED)
            gemm_load_step(A, Wt, m0, n0, k0 + 64, ln, half, a0, b0);
#pragma unroll
        for (int mt = 0; mt < 2; ++mt)
#pragma unroll
            for (int nt = 0; nt < 4; ++nt)
                acc[mt][nt] = wmma_bf16(a1[mt], b1[nt], acc[mt][nt]);
    }
}

// ---------------- QKV projection GEMM ----------------
// 128-thread blocks: 4 waves, each computing a 32x64 tile.
// Stores Q,K as [BH,T,64] bf16 and V transposed [BH,64,T].
__global__ void __launch_bounds__(128, 1)
qkv_gemm(const __bf16* __restrict__ xb,
         const __bf16* __restrict__ Wqt, const __bf16* __restrict__ Wkt,
         const __bf16* __restrict__ Wvt,
         const float* __restrict__ bq, const float* __restrict__ bk,
         const float* __restrict__ bv,
         __bf16* __restrict__ Qb, __bf16* __restrict__ Kb,
         __bf16* __restrict__ Vt) {
    const int which = blockIdx.z;
    const __bf16* Wt   = (which == 0) ? Wqt : (which == 1) ? Wkt : Wvt;
    const float*  bias = (which == 0) ? bq  : (which == 1) ? bk  : bv;
    const int wid = threadIdx.x >> 5;
    const int n0 = blockIdx.x * 64;
    const int m0 = blockIdx.y * 128 + wid * 32;
    const int lane = threadIdx.x & 31, ln = lane & 15, half = lane >> 4;

    v8f acc[2][4] = {};
    gemm_32x64(xb, Wt, m0, n0, ln, half, acc);

    // C layout: N = ln (fixed per lane), M = v + 8*half
#pragma unroll
    for (int mt = 0; mt < 2; ++mt)
#pragma unroll
        for (int nt = 0; nt < 4; ++nt) {
            const int n = n0 + nt * 16 + ln;
            const int h = n >> 6, d = n & 63;
            const float bb = bias[n];
#pragma unroll
            for (int v = 0; v < 8; ++v) {
                const int m  = m0 + mt * 16 + v + 8 * half;
                const int b_ = m / SEQ, t = m % SEQ;
                const int bh = b_ * NHEAD + h;
                const float val = acc[mt][nt][v] + bb;
                if (which == 2)      Vt[(bh * HDIM + d) * SEQ + t] = (__bf16)val;
                else if (which == 1) Kb[(bh * SEQ + t) * HDIM + d] = (__bf16)val;
                else                 Qb[(bh * SEQ + t) * HDIM + d] = (__bf16)val;
            }
        }
}

// ---------------- flash attention (transposed-softmax, K double-buffered) ----------------
__device__ __forceinline__ void attn_load_k(const __bf16* __restrict__ Kh, int kbase,
                                            int ln, int half, v16bf ka[2][2]) {
#pragma unroll
    for (int st = 0; st < 2; ++st)
#pragma unroll
        for (int dd = 0; dd < 2; ++dd) {
            const __bf16* p = Kh + (kbase + st * 16 + ln) * HDIM + dd * 32 + (half ? 8 : 0);
            ka[st][dd] = load_a16(p, p + 16);   // A: M=key=ln, K=d
        }
}

__device__ __forceinline__ void attn_load_v(const __bf16* __restrict__ Vh, int kbase,
                                            int ln, int half, v16bf va[4]) {
#pragma unroll
    for (int mt = 0; mt < 4; ++mt) {
        const __bf16* p = Vh + (mt * 16 + ln) * SEQ + kbase + (half ? 16 : 0);
        va[mt] = load_a16(p, p + 8);            // A: M=d, K=key
    }
}

__device__ __forceinline__ void attn_step(int kbase, int q, int half,
                                          const v16bf bqt[2], const v16bf ka[2][2],
                                          const v16bf va[4], v8f acc[4],
                                          float& m_run, float& l_run) {
    // S^T sub-tiles: 16 keys x 16 queries each
    v8f s[2];
#pragma unroll
    for (int st = 0; st < 2; ++st) {
        v8f c = {};
#pragma unroll
        for (int dd = 0; dd < 2; ++dd)
            c = wmma_bf16(ka[st][dd], bqt[dd], c);
        s[st] = c;
    }
    // scale (1/sqrt(64)=0.125), causal mask, local row max
    float mloc = -1e30f;
#pragma unroll
    for (int st = 0; st < 2; ++st)
#pragma unroll
        for (int v = 0; v < 8; ++v) {
            const int key = kbase + st * 16 + v + 8 * half;
            const float sv = (key <= q) ? s[st][v] * 0.125f : -1e30f;
            s[st][v] = sv;
            mloc = fmaxf(mloc, sv);
        }
    mloc = fmaxf(mloc, __shfl_xor(mloc, 16));       // combine half-waves
    const float m_new = fmaxf(m_run, mloc);
    const float alpha = __expf(m_run - m_new);
    float psum = 0.f;
#pragma unroll
    for (int st = 0; st < 2; ++st)
#pragma unroll
        for (int v = 0; v < 8; ++v) {
            const float p = __expf(s[st][v] - m_new);
            s[st][v] = p;
            psum += p;
        }
    psum += __shfl_xor(psum, 16);
    l_run = l_run * alpha + psum;
#pragma unroll
    for (int mt = 0; mt < 4; ++mt)
#pragma unroll
        for (int v = 0; v < 8; ++v) acc[mt][v] *= alpha;

    // Repack P^T (C-layout) into WMMA B-layout bf16 (K=keys, N=queries).
    v16bf pb;
#pragma unroll
    for (int j = 0; j < 8; ++j) {
        const int i = (j < 4) ? 2 * j : 2 * j - 8;
        const float s0a = s[0][i], s0b = s[0][i + 1];
        const float s1a = s[1][i], s1b = s[1][i + 1];
        const float r0a = __shfl_xor(s0a, 16), r0b = __shfl_xor(s0b, 16);
        const float r1a = __shfl_xor(s1a, 16), r1b = __shfl_xor(s1b, 16);
        float va_, vb_;
        if (j < 4) { va_ = half ? r1a : s0a; vb_ = half ? r1b : s0b; }
        else       { va_ = half ? s1a : r0a; vb_ = half ? s1b : r0b; }
        pb[2 * j]     = (__bf16)va_;
        pb[2 * j + 1] = (__bf16)vb_;
    }
    // O^T += V^T . P^T
#pragma unroll
    for (int mt = 0; mt < 4; ++mt)
        acc[mt] = wmma_bf16(va[mt], pb, acc[mt]);
    m_run = m_new;
}

__global__ void __launch_bounds__(128, 1)
attn_kernel(const __bf16* __restrict__ Qb, const __bf16* __restrict__ Kb,
            const __bf16* __restrict__ Vt, __bf16* __restrict__ attnb) {
    const int wid = threadIdx.x >> 5;
    const int qb = blockIdx.x * 4 + wid;     // 16-query block per wave
    const int bh = blockIdx.y;               // b*NHEAD + h
    const int lane = threadIdx.x & 31, ln = lane & 15, half = lane >> 4;
    const __bf16* Qh = Qb + (size_t)bh * SEQ * HDIM;
    const __bf16* Kh = Kb + (size_t)bh * SEQ * HDIM;
    const __bf16* Vh = Vt + (size_t)bh * HDIM * SEQ;
    const int q = qb * 16 + ln;              // this lane's query index

    // Preload B-tiles of Q^T for d0 = 0, 32 (B: N=query=ln, K=d table order)
    v16bf bqt[2];
#pragma unroll
    for (int dd = 0; dd < 2; ++dd) {
        const __bf16* p = Qh + q * HDIM + dd * 32 + (half ? 8 : 0);
        bqt[dd] = load_a16(p, p + 16);
    }

    v8f acc[4] = {};                         // O^T: 4 d-tiles of 16, N=query=ln
    float m_run = -1e30f, l_run = 0.f;
    const int nsteps = (16 * (qb + 1) + 31) >> 5;   // 32 keys per step, causal

    // Pipeline: K tiles (consumed immediately) are double-buffered across steps;
    // V tiles (consumed after the softmax VALU chain) are issued at step start.
    v16bf kaA[2][2], kaB[2][2], va[4];
    attn_load_k(Kh, 0, ln, half, kaA);
    int step = 0;
    while (true) {
        attn_load_v(Vh, step * 32, ln, half, va);
        if (step + 1 < nsteps) attn_load_k(Kh, (step + 1) * 32, ln, half, kaB);
        attn_step(step * 32, q, half, bqt, kaA, va, acc, m_run, l_run);
        ++step; if (step >= nsteps) break;
        attn_load_v(Vh, step * 32, ln, half, va);
        if (step + 1 < nsteps) attn_load_k(Kh, (step + 1) * 32, ln, half, kaA);
        attn_step(step * 32, q, half, bqt, kaB, va, acc, m_run, l_run);
        ++step; if (step >= nsteps) break;
    }

    const float inv = 1.f / l_run;
    const int b_ = bh >> 4, h = bh & 15;
    const int row = b_ * SEQ + q;
#pragma unroll
    for (int mt = 0; mt < 4; ++mt)
#pragma unroll
        for (int v = 0; v < 8; ++v) {
            const int d = mt * 16 + v + 8 * half;
            attnb[(size_t)row * EMBED + h * HDIM + d] = (__bf16)(acc[mt][v] * inv);
        }
}

// ---------------- output projection GEMM (bf16 -> fp32 out) ----------------
__global__ void __launch_bounds__(128, 1)
out_gemm(const __bf16* __restrict__ attnb, const __bf16* __restrict__ Wot,
         const float* __restrict__ bo, float* __restrict__ out) {
    const int wid = threadIdx.x >> 5;
    const int n0 = blockIdx.x * 64;
    const int m0 = blockIdx.y * 128 + wid * 32;
    const int lane = threadIdx.x & 31, ln = lane & 15, half = lane >> 4;

    v8f acc[2][4] = {};
    gemm_32x64(attnb, Wot, m0, n0, ln, half, acc);

#pragma unroll
    for (int mt = 0; mt < 2; ++mt)
#pragma unroll
        for (int nt = 0; nt < 4; ++nt) {
            const int n = n0 + nt * 16 + ln;
            const float bb = bo[n];
#pragma unroll
            for (int v = 0; v < 8; ++v) {
                const int m = m0 + mt * 16 + v + 8 * half;
                out[(size_t)m * EMBED + n] = acc[mt][nt][v] + bb;
            }
        }
}

// ---------------- launcher ----------------
extern "C" void kernel_launch(void* const* d_in, const int* in_sizes, int n_in,
                              void* d_out, int out_size, void* d_ws, size_t ws_size,
                              hipStream_t stream) {
    const float* x   = (const float*)d_in[0];
    const float* Wq  = (const float*)d_in[1];
    const float* bq  = (const float*)d_in[2];
    const float* Wk  = (const float*)d_in[3];
    const float* bk  = (const float*)d_in[4];
    const float* Wv  = (const float*)d_in[5];
    const float* bv  = (const float*)d_in[6];
    const float* Wo  = (const float*)d_in[7];
    const float* bo  = (const float*)d_in[8];
    float* out = (float*)d_out;

    char* w = (char*)d_ws;
    size_t off = 0;
    const size_t sz_x  = (size_t)MROWS * EMBED * sizeof(__bf16);   // 8 MB
    const size_t sz_w  = (size_t)EMBED * EMBED * sizeof(__bf16);   // 2 MB
    __bf16* xb    = (__bf16*)(w + off); off += sz_x;
    __bf16* Wqt   = (__bf16*)(w + off); off += sz_w;
    __bf16* Wkt   = (__bf16*)(w + off); off += sz_w;
    __bf16* Wvt   = (__bf16*)(w + off); off += sz_w;
    __bf16* Wot   = (__bf16*)(w + off); off += sz_w;
    __bf16* Qb    = (__bf16*)(w + off); off += sz_x;
    __bf16* Kb    = (__bf16*)(w + off); off += sz_x;
    __bf16* Vtm   = (__bf16*)(w + off); off += sz_x;
    __bf16* attnb = (__bf16*)(w + off); off += sz_x;               // total 48 MB

    const int nelem = MROWS * EMBED;
    cvt_f32_bf16<<<nelem / 256, 256, 0, stream>>>(x, xb, nelem);
    dim3 tgrid(EMBED / 32, EMBED / 32), tblk(32, 8);
    transpose_cvt<<<tgrid, tblk, 0, stream>>>(Wq, Wqt);
    transpose_cvt<<<tgrid, tblk, 0, stream>>>(Wk, Wkt);
    transpose_cvt<<<tgrid, tblk, 0, stream>>>(Wv, Wvt);
    transpose_cvt<<<tgrid, tblk, 0, stream>>>(Wo, Wot);

    qkv_gemm<<<dim3(EMBED / 64, MROWS / 128, 3), 128, 0, stream>>>(
        xb, Wqt, Wkt, Wvt, bq, bk, bv, Qb, Kb, Vtm);

    attn_kernel<<<dim3(SEQ / 16 / 4, BATCH * NHEAD), 128, 0, stream>>>(Qb, Kb, Vtm, attnb);

    out_gemm<<<dim3(EMBED / 64, MROWS / 128), 128, 0, stream>>>(attnb, Wot, bo, out);
}
